// MultiheadAttention_80942953660632
// MI455X (gfx1250) — matrix operand order repaired and checked by reading, not compiled, not run on hardware
//
#include <hip/hip_runtime.h>

// ---------------------------------------------------------------------------
// MultiheadAttention ("product" bilinear attention) for gfx1250 / MI455X.
// GEMM stages: v_wmma_f32_16x16x32_f16 (f32 accumulate), wave32.
// f16 tile staging: GLOBAL_LOAD_ASYNC_TO_LDS_B128 (ASYNCcnt) via inline asm.
// ---------------------------------------------------------------------------

typedef __attribute__((ext_vector_type(16))) _Float16 v16h;
typedef __attribute__((ext_vector_type(8)))  _Float16 v8h;
typedef __attribute__((ext_vector_type(8)))  float    v8f;

constexpr int E_  = 1024;  // embed dim
constexpr int H_  = 16;    // heads
constexpr int D_  = 64;    // head dim
constexpr int L_  = 1024;  // query len
constexpr int S_  = 1024;  // key len
constexpr int NB_ = 4;     // batch
constexpr int MR_ = L_ * NB_;  // 4096 flattened rows

__device__ __forceinline__ v8f vzero8() {
  v8f z = {0.f, 0.f, 0.f, 0.f, 0.f, 0.f, 0.f, 0.f};
  return z;
}

// Async 16-byte copy global -> LDS (per-lane), tracked by ASYNCcnt.
// LDS operand is the workgroup-relative LDS byte address (low 32 bits of the
// generic pointer per the CDNA5 aperture mapping).
__device__ __forceinline__ void async_b128(const void* g, void* l) {
  unsigned loff = (unsigned)(uintptr_t)l;
  unsigned long long ga = (unsigned long long)(uintptr_t)g;
  asm volatile("global_load_async_to_lds_b128 %0, %1, off"
               :: "v"(loff), "v"(ga)
               : "memory");
}
__device__ __forceinline__ void async_wait0() {
  asm volatile("s_wait_asynccnt 0" ::: "memory");
}

// A-fragment (16x32 f16, MxK). Per CDNA5 ISA: lanes 0-15 -> M=lane, K 0-7 in
// VGPR0-3 and K 16-23 in VGPR4-7; lanes 16-31 -> K 8-15 / 24-31.
__device__ __forceinline__ v16h lds_frag_a(const _Float16* t, int ld, int m0, int k0) {
  const int lane = threadIdx.x & 31;
  const int m = m0 + (lane & 15);
  const int g = lane >> 4;
  const _Float16* p = t + m * ld + k0 + g * 8;
  v8h lo = *(const v8h*)(p);
  v8h hi = *(const v8h*)(p + 16);
  v16h a;
#pragma unroll
  for (int i = 0; i < 8; ++i) { a[i] = lo[i]; a[8 + i] = hi[i]; }
  return a;
}

// B-fragment (32x16 f16, KxN). Lanes 0-15 hold K=0-7,16-23 of col n=lane,
// lanes 16-31 hold K=8-15,24-31. LDS tile stored K-contiguous as [n][k].
__device__ __forceinline__ v16h lds_frag_b(const _Float16* t, int ld, int n0, int k0) {
  const int lane = threadIdx.x & 31;
  const int n = n0 + (lane & 15);
  const int g = lane >> 4;
  const _Float16* p = t + n * ld + k0 + g * 8;
  v8h lo = *(const v8h*)(p);
  v8h hi = *(const v8h*)(p + 16);
  v16h b;
#pragma unroll
  for (int i = 0; i < 8; ++i) { b[i] = lo[i]; b[8 + i] = hi[i]; }
  return b;
}

__device__ __forceinline__ v8f wmma16(v16h a, v16h b, v8f c) {
  return __builtin_amdgcn_wmma_f32_16x16x32_f16(false, a, false, b, (short)0, c,
                                                false, false);
}

// ---------------------------------------------------------------------------
// Kernel 1: in-projection  C[r,o] = sum_i A[r,i]*W[o,i] + bias[o]
// fp32 inputs -> f16 conversion fused in staging (two-phase: all loads in
// flight before first consume). Epilogue de-interleaves heads (e = d*H + h):
//   vmode=0: Out[h][n][row][d]   (Q, K)
//   vmode=1: Out[h][n][d][row]   (V, transposed so PV B-operand is K-contig)
// ---------------------------------------------------------------------------
__global__ __launch_bounds__(256) void k_inproj(const float* __restrict__ A,
                                                const float* __restrict__ Wt,
                                                const float* __restrict__ bias,
                                                _Float16* __restrict__ Out,
                                                int Lrows, int vmode) {
  constexpr int BM = 128, BN = 128, BK = 64, LDT = BK + 8;
  __shared__ _Float16 As[BM * LDT];
  __shared__ _Float16 Bs[BN * LDT];
  const int tid = threadIdx.x;
  const int bm = blockIdx.x * BM;
  const int bn = blockIdx.y * BN;
  const int wid = tid >> 5, lane = tid & 31;
  const int wm = (wid >> 2) * 64;
  const int wn = (wid & 3) * 32;
  v8f acc[4][2];
#pragma unroll
  for (int i = 0; i < 4; ++i)
#pragma unroll
    for (int j = 0; j < 2; ++j) acc[i][j] = vzero8();

  const int srow = tid >> 4, sc4 = (tid & 15) * 4;  // this thread's chunk base
  for (int k0 = 0; k0 < E_; k0 += BK) {
    float4 ra[8], rb[8];
#pragma unroll
    for (int j = 0; j < 8; ++j) {  // phase 1: all 16 loads in flight
      int row = srow + 16 * j;
      ra[j] = *(const float4*)(A  + (size_t)(bm + row) * E_ + k0 + sc4);
      rb[j] = *(const float4*)(Wt + (size_t)(bn + row) * E_ + k0 + sc4);
    }
#pragma unroll
    for (int j = 0; j < 8; ++j) {  // phase 2: convert + LDS store
      int row = srow + 16 * j;
      _Float16* da = As + row * LDT + sc4;
      da[0] = (_Float16)ra[j].x; da[1] = (_Float16)ra[j].y;
      da[2] = (_Float16)ra[j].z; da[3] = (_Float16)ra[j].w;
      _Float16* db = Bs + row * LDT + sc4;
      db[0] = (_Float16)rb[j].x; db[1] = (_Float16)rb[j].y;
      db[2] = (_Float16)rb[j].z; db[3] = (_Float16)rb[j].w;
    }
    __syncthreads();
#pragma unroll
    for (int kk = 0; kk < BK; kk += 32) {
      v16h b0 = lds_frag_b(Bs, LDT, wn, kk);
      v16h b1 = lds_frag_b(Bs, LDT, wn + 16, kk);
#pragma unroll
      for (int mi = 0; mi < 4; ++mi) {
        v16h a = lds_frag_a(As, LDT, wm + mi * 16, kk);
        acc[mi][0] = wmma16(a, b0, acc[mi][0]);
        acc[mi][1] = wmma16(a, b1, acc[mi][1]);
      }
    }
    __syncthreads();
  }
  const int g = lane >> 4, cl = lane & 15;
#pragma unroll
  for (int mi = 0; mi < 4; ++mi)
#pragma unroll
    for (int ni = 0; ni < 2; ++ni) {
      int gcol = bn + wn + ni * 16 + cl;
      int d = gcol >> 4, h = gcol & 15;
      float bv = bias[gcol];
#pragma unroll
      for (int v = 0; v < 8; ++v) {
        int grow = bm + wm + mi * 16 + g * 8 + v;
        int l = grow >> 2, nb = grow & 3;  // rows are [L][N] flattened
        float val = acc[mi][ni][v] + bv;
        size_t o = vmode ? (((size_t)(h * NB_ + nb) * D_ + d) * Lrows + l)
                         : (((size_t)(h * NB_ + nb) * Lrows + l) * D_ + d);
        Out[o] = (_Float16)val;
      }
    }
}

// ---------------------------------------------------------------------------
// Kernel 2: per-head transform  Qt[r,i] = sum_o Qh[r,o] * attn_W[o,i,h] / 32
// ---------------------------------------------------------------------------
__global__ __launch_bounds__(256) void k_qt(const _Float16* __restrict__ Qh,
                                            const float* __restrict__ attnW,
                                            _Float16* __restrict__ Qt) {
  constexpr int BM = 128, LDT = D_ + 8;
  __shared__ _Float16 As[BM * LDT];
  __shared__ _Float16 Bs[D_ * LDT];
  const int tid = threadIdx.x;
  const int h = blockIdx.y;
  const int bm = blockIdx.x * BM;
  const _Float16* Ab = Qh + (size_t)h * (NB_ * L_ * D_);
#pragma unroll
  for (int j = 0; j < 4; ++j) {  // A: 128x64 f16 (async copy)
    int idx = tid + 256 * j;
    int row = idx >> 3, c8 = (idx & 7) * 8;
    async_b128(Ab + (size_t)(bm + row) * D_ + c8, As + row * LDT + c8);
  }
#pragma unroll
  for (int j = 0; j < 16; ++j) {  // B transposed+scaled: Bs[i][o]=W[o,i,h]/32
    int idx = tid + 256 * j;
    int o = idx >> 6, i = idx & 63;
    Bs[i * LDT + o] = (_Float16)(attnW[(size_t)(o * D_ + i) * H_ + h] * 0.03125f);
  }
  async_wait0();
  __syncthreads();
  const int wid = tid >> 5, lane = tid & 31;
  const int wm = (wid >> 1) * 32, wn = (wid & 1) * 32;
  v8f acc[2][2] = {vzero8(), vzero8(), vzero8(), vzero8()};
#pragma unroll
  for (int kk = 0; kk < D_; kk += 32) {
    v16h b0 = lds_frag_b(Bs, LDT, wn, kk);
    v16h b1 = lds_frag_b(Bs, LDT, wn + 16, kk);
#pragma unroll
    for (int mi = 0; mi < 2; ++mi) {
      v16h a = lds_frag_a(As, LDT, wm + mi * 16, kk);
      acc[mi][0] = wmma16(a, b0, acc[mi][0]);
      acc[mi][1] = wmma16(a, b1, acc[mi][1]);
    }
  }
  _Float16* Ob = Qt + (size_t)h * (NB_ * L_ * D_);
  const int g = lane >> 4, cl = lane & 15;
#pragma unroll
  for (int mi = 0; mi < 2; ++mi)
#pragma unroll
    for (int ni = 0; ni < 2; ++ni) {
      int col = wn + ni * 16 + cl;
#pragma unroll
      for (int v = 0; v < 8; ++v) {
        int row = bm + wm + mi * 16 + g * 8 + v;
        Ob[(size_t)row * D_ + col] = (_Float16)acc[mi][ni][v];
      }
    }
}

// ---------------------------------------------------------------------------
// Kernel 3: softmax row stats (max, sum-exp) over S, per (h, n, 64 l-rows).
// WMMA score tiles + cross-lane shuffle reductions (online softmax).
// ---------------------------------------------------------------------------
__global__ __launch_bounds__(256) void k_stats(const _Float16* __restrict__ Qt,
                                               const _Float16* __restrict__ Kh,
                                               float* __restrict__ Mrow,
                                               float* __restrict__ Zrow) {
  constexpr int LDT = D_ + 8;
  __shared__ _Float16 Qs[64 * LDT];
  __shared__ _Float16 Ks[128 * LDT];
  __shared__ float SM[64 * 4];
  __shared__ float SZ[64 * 4];
  const int tid = threadIdx.x;
  const int h = blockIdx.z, n = blockIdx.y;
  const int l0 = blockIdx.x * 64;
  const _Float16* Qb = Qt + ((size_t)(h * NB_ + n) * L_ + l0) * D_;
  const _Float16* Kb = Kh + (size_t)(h * NB_ + n) * S_ * D_;
#pragma unroll
  for (int j = 0; j < 2; ++j) {
    int idx = tid + 256 * j;
    int row = idx >> 3, c8 = (idx & 7) * 8;
    async_b128(Qb + (size_t)row * D_ + c8, Qs + row * LDT + c8);
  }
  const int wid = tid >> 5, lane = tid & 31;
  const int wm = (wid >> 2) * 32, wn = (wid & 3) * 32;
  const int g = lane >> 4;
  float rm[2][8], rz[2][8];
#pragma unroll
  for (int mi = 0; mi < 2; ++mi)
#pragma unroll
    for (int v = 0; v < 8; ++v) { rm[mi][v] = -1e30f; rz[mi][v] = 0.f; }

  for (int s0 = 0; s0 < S_; s0 += 128) {
    __syncthreads();
#pragma unroll
    for (int j = 0; j < 4; ++j) {
      int idx = tid + 256 * j;
      int row = idx >> 3, c8 = (idx & 7) * 8;
      async_b128(Kb + (size_t)(s0 + row) * D_ + c8, Ks + row * LDT + c8);
    }
    async_wait0();
    __syncthreads();
    v8f acc[2][2] = {vzero8(), vzero8(), vzero8(), vzero8()};
#pragma unroll
    for (int kk = 0; kk < D_; kk += 32) {
      v16h b0 = lds_frag_b(Ks, LDT, wn, kk);
      v16h b1 = lds_frag_b(Ks, LDT, wn + 16, kk);
#pragma unroll
      for (int mi = 0; mi < 2; ++mi) {
        v16h a = lds_frag_a(Qs, LDT, wm + mi * 16, kk);
        acc[mi][0] = wmma16(a, b0, acc[mi][0]);
        acc[mi][1] = wmma16(a, b1, acc[mi][1]);
      }
    }
#pragma unroll
    for (int mi = 0; mi < 2; ++mi)
#pragma unroll
      for (int v = 0; v < 8; ++v) {
        float cm = fmaxf(acc[mi][0][v], acc[mi][1][v]);
#pragma unroll
        for (int msk = 1; msk < 16; msk <<= 1)
          cm = fmaxf(cm, __shfl_xor(cm, msk, 32));
        float cz = __expf(acc[mi][0][v] - cm) + __expf(acc[mi][1][v] - cm);
#pragma unroll
        for (int msk = 1; msk < 16; msk <<= 1) cz += __shfl_xor(cz, msk, 32);
        float nm = fmaxf(rm[mi][v], cm);
        rz[mi][v] = rz[mi][v] * __expf(rm[mi][v] - nm) + cz * __expf(cm - nm);
        rm[mi][v] = nm;
      }
  }
  __syncthreads();
  if ((lane & 15) == 0) {
#pragma unroll
    for (int mi = 0; mi < 2; ++mi)
#pragma unroll
      for (int v = 0; v < 8; ++v) {
        int row = wm + mi * 16 + g * 8 + v;
        SM[row * 4 + (wid & 3)] = rm[mi][v];
        SZ[row * 4 + (wid & 3)] = rz[mi][v];
      }
  }
  __syncthreads();
  if (tid < 64) {
    float gm = -1e30f;
#pragma unroll
    for (int w = 0; w < 4; ++w) gm = fmaxf(gm, SM[tid * 4 + w]);
    float gz = 0.f;
#pragma unroll
    for (int w = 0; w < 4; ++w) gz += SZ[tid * 4 + w] * __expf(SM[tid * 4 + w] - gm);
    size_t o = (size_t)(h * NB_ + n) * L_ + l0 + tid;
    Mrow[o] = gm;
    Zrow[o] = gz;
  }
}

// ---------------------------------------------------------------------------
// Kernel 4: recompute scores, normalize to f16 probs in LDS, P x V via WMMA.
// Writes O (f16) in [L][N][E] with e = d*H + h.
// ---------------------------------------------------------------------------
__global__ __launch_bounds__(256) void k_pv(const _Float16* __restrict__ Qt,
                                            const _Float16* __restrict__ Kh,
                                            const _Float16* __restrict__ Vt,
                                            const float* __restrict__ Mrow,
                                            const float* __restrict__ Zrow,
                                            _Float16* __restrict__ O16) {
  constexpr int LDK = D_ + 8, LDS2 = 128 + 8;
  __shared__ _Float16 Qs[64 * LDK];
  __shared__ _Float16 Ks[128 * LDK];
  __shared__ _Float16 Vs[64 * LDS2];   // [d][s] (V pre-transposed in global)
  __shared__ _Float16 Ps[64 * LDS2];   // [l][s]
  __shared__ float rM[64], rZi[64];
  const int tid = threadIdx.x;
  const int h = blockIdx.z, n = blockIdx.y;
  const int l0 = blockIdx.x * 64;
  const _Float16* Qb = Qt + ((size_t)(h * NB_ + n) * L_ + l0) * D_;
  const _Float16* Kb = Kh + (size_t)(h * NB_ + n) * S_ * D_;
  const _Float16* Vb = Vt + (size_t)(h * NB_ + n) * D_ * S_;
#pragma unroll
  for (int j = 0; j < 2; ++j) {
    int idx = tid + 256 * j;
    int row = idx >> 3, c8 = (idx & 7) * 8;
    async_b128(Qb + (size_t)row * D_ + c8, Qs + row * LDK + c8);
  }
  if (tid < 64) {
    size_t o = (size_t)(h * NB_ + n) * L_ + l0 + tid;
    rM[tid] = Mrow[o];
    rZi[tid] = 1.0f / Zrow[o];
  }
  const int wid = tid >> 5, lane = tid & 31;
  const int g = lane >> 4, cl = lane & 15;
  const int wmS = (wid >> 2) * 32, wnS = (wid & 3) * 32;  // score phase 64x128
  const int wmP = (wid >> 2) * 32, wnP = (wid & 3) * 16;  // pv phase 64x64
  v8f accO[2] = {vzero8(), vzero8()};

  for (int s0 = 0; s0 < S_; s0 += 128) {
    __syncthreads();
#pragma unroll
    for (int j = 0; j < 4; ++j) {  // K tile [s][d]
      int idx = tid + 256 * j;
      int row = idx >> 3, c8 = (idx & 7) * 8;
      async_b128(Kb + (size_t)(s0 + row) * D_ + c8, Ks + row * LDK + c8);
    }
#pragma unroll
    for (int j = 0; j < 4; ++j) {  // V tile [d][s]
      int idx = tid + 256 * j;
      int row = idx >> 4, c8 = (idx & 15) * 8;
      async_b128(Vb + (size_t)row * S_ + s0 + c8, Vs + row * LDS2 + c8);
    }
    async_wait0();
    __syncthreads();
    v8f acc[2][2] = {vzero8(), vzero8(), vzero8(), vzero8()};
#pragma unroll
    for (int kk = 0; kk < D_; kk += 32) {
      v16h b0 = lds_frag_b(Ks, LDK, wnS, kk);
      v16h b1 = lds_frag_b(Ks, LDK, wnS + 16, kk);
#pragma unroll
      for (int mi = 0; mi < 2; ++mi) {
        v16h a = lds_frag_a(Qs, LDK, wmS + mi * 16, kk);
        acc[mi][0] = wmma16(a, b0, acc[mi][0]);
        acc[mi][1] = wmma16(a, b1, acc[mi][1]);
      }
    }
#pragma unroll
    for (int mi = 0; mi < 2; ++mi)
#pragma unroll
      for (int ni = 0; ni < 2; ++ni)
#pragma unroll
        for (int v = 0; v < 8; ++v) {
          int row = wmS + mi * 16 + g * 8 + v;
          float p = __expf(acc[mi][ni][v] - rM[row]) * rZi[row];
          Ps[row * LDS2 + wnS + ni * 16 + cl] = (_Float16)p;
        }
    __syncthreads();
#pragma unroll
    for (int kk = 0; kk < 128; kk += 32) {
      v16h b = lds_frag_b(Vs, LDS2, wnP, kk);
#pragma unroll
      for (int mi = 0; mi < 2; ++mi) {
        v16h a = lds_frag_a(Ps, LDS2, wmP + mi * 16, kk);
        accO[mi] = wmma16(a, b, accO[mi]);
      }
    }
  }
#pragma unroll
  for (int mi = 0; mi < 2; ++mi)
#pragma unroll
    for (int v = 0; v < 8; ++v) {
      int row = wmP + mi * 16 + g * 8 + v;
      int d = wnP + cl;
      O16[((size_t)(l0 + row) * NB_ + n) * E_ + d * H_ + h] =
          (_Float16)accO[mi][v];
    }
}

// ---------------------------------------------------------------------------
// Kernel 5: attention map  Wmap[l,s,n] = (1/H) * sum_h exp(score-m)/Z.
// Loops all heads per (n, l-tile, s-tile); no atomics -> deterministic.
// ---------------------------------------------------------------------------
__global__ __launch_bounds__(256) void k_wmap(const _Float16* __restrict__ Qt,
                                              const _Float16* __restrict__ Kh,
                                              const float* __restrict__ Mrow,
                                              const float* __restrict__ Zrow,
                                              float* __restrict__ Wout) {
  constexpr int LDT = D_ + 8;
  __shared__ _Float16 Qs[64 * LDT];
  __shared__ _Float16 Ks[128 * LDT];
  __shared__ float rM[64], rZi[64];
  const int tid = threadIdx.x;
  const int n = blockIdx.z;
  const int l0 = blockIdx.y * 64;
  const int s0 = blockIdx.x * 128;
  const int wid = tid >> 5, lane = tid & 31;
  const int wm = (wid >> 2) * 32, wn = (wid & 3) * 32;
  const int g = lane >> 4, cl = lane & 15;
  v8f Wacc[2][2] = {vzero8(), vzero8(), vzero8(), vzero8()};

  for (int h = 0; h < H_; ++h) {
    __syncthreads();
    const _Float16* Qb = Qt + ((size_t)(h * NB_ + n) * L_ + l0) * D_;
    const _Float16* Kb = Kh + (size_t)(h * NB_ + n) * S_ * D_;
#pragma unroll
    for (int j = 0; j < 2; ++j) {
      int idx = tid + 256 * j;
      int row = idx >> 3, c8 = (idx & 7) * 8;
      async_b128(Qb + (size_t)row * D_ + c8, Qs + row * LDT + c8);
    }
#pragma unroll
    for (int j = 0; j < 4; ++j) {
      int idx = tid + 256 * j;
      int row = idx >> 3, c8 = (idx & 7) * 8;
      async_b128(Kb + (size_t)(s0 + row) * D_ + c8, Ks + row * LDT + c8);
    }
    if (tid < 64) {
      size_t o = (size_t)(h * NB_ + n) * L_ + l0 + tid;
      rM[tid] = Mrow[o];
      rZi[tid] = 1.0f / Zrow[o];
    }
    async_wait0();
    __syncthreads();
    v8f acc[2][2] = {vzero8(), vzero8(), vzero8(), vzero8()};
#pragma unroll
    for (int kk = 0; kk < D_; kk += 32) {
      v16h b0 = lds_frag_b(Ks, LDT, wn, kk);
      v16h b1 = lds_frag_b(Ks, LDT, wn + 16, kk);
#pragma unroll
      for (int mi = 0; mi < 2; ++mi) {
        v16h a = lds_frag_a(Qs, LDT, wm + mi * 16, kk);
        acc[mi][0] = wmma16(a, b0, acc[mi][0]);
        acc[mi][1] = wmma16(a, b1, acc[mi][1]);
      }
    }
#pragma unroll
    for (int mi = 0; mi < 2; ++mi)
#pragma unroll
      for (int ni = 0; ni < 2; ++ni)
#pragma unroll
        for (int v = 0; v < 8; ++v) {
          int row = wm + mi * 16 + g * 8 + v;
          Wacc[mi][ni][v] += __expf(acc[mi][ni][v] - rM[row]) * rZi[row];
        }
  }
#pragma unroll
  for (int mi = 0; mi < 2; ++mi)
#pragma unroll
    for (int ni = 0; ni < 2; ++ni)
#pragma unroll
      for (int v = 0; v < 8; ++v) {
        int l = l0 + wm + mi * 16 + g * 8 + v;
        int s = s0 + wn + ni * 16 + cl;
        Wout[((size_t)l * S_ + s) * NB_ + n] = Wacc[mi][ni][v] * (1.0f / H_);
      }
}

// ---------------------------------------------------------------------------
// Kernel 6: out-projection  out[r,o] = sum_e O[r,e]*out_w[e,o] + out_bias[o]
// A (f16) staged async; B (fp32) transposed+converted via registers.
// ---------------------------------------------------------------------------
__global__ __launch_bounds__(256) void k_outproj(const _Float16* __restrict__ A,
                                                 const float* __restrict__ Wb,
                                                 const float* __restrict__ bias,
                                                 float* __restrict__ Out) {
  constexpr int BM = 128, BN = 128, BK = 64, LDT = BK + 8;
  __shared__ _Float16 As[BM * LDT];
  __shared__ _Float16 Bs[BN * LDT];
  const int tid = threadIdx.x;
  const int bm = blockIdx.x * BM;
  const int bn = blockIdx.y * BN;
  const int wid = tid >> 5, lane = tid & 31;
  const int wm = (wid >> 2) * 64, wn = (wid & 3) * 32;
  v8f acc[4][2];
#pragma unroll
  for (int i = 0; i < 4; ++i)
#pragma unroll
    for (int j = 0; j < 2; ++j) acc[i][j] = vzero8();

  const int be = tid >> 5 /*unused*/;
  (void)be;
  for (int k0 = 0; k0 < E_; k0 += BK) {
#pragma unroll
    for (int j = 0; j < 4; ++j) {  // A (f16): 128x64 async copy
      int idx = tid + 256 * j;
      int row = idx >> 3, c8 = (idx & 7) * 8;
      async_b128(A + (size_t)(bm + row) * E_ + k0 + c8, As + row * LDT + c8);
    }
    float4 rb[8];
#pragma unroll
    for (int j = 0; j < 8; ++j) {  // B loads first (all in flight)
      int idx = tid + 256 * j;
      int e = idx >> 5, c4 = (idx & 31) * 4;
      rb[j] = *(const float4*)(Wb + (size_t)(k0 + e) * E_ + bn + c4);
    }
#pragma unroll
    for (int j = 0; j < 8; ++j) {  // B: transpose + convert -> Bs[o][e]
      int idx = tid + 256 * j;
      int e = idx >> 5, c4 = (idx & 31) * 4;
      Bs[(c4 + 0) * LDT + e] = (_Float16)rb[j].x;
      Bs[(c4 + 1) * LDT + e] = (_Float16)rb[j].y;
      Bs[(c4 + 2) * LDT + e] = (_Float16)rb[j].z;
      Bs[(c4 + 3) * LDT + e] = (_Float16)rb[j].w;
    }
    async_wait0();
    __syncthreads();
#pragma unroll
    for (int kk = 0; kk < BK; kk += 32) {
      v16h b0 = lds_frag_b(Bs, LDT, wn, kk);
      v16h b1 = lds_frag_b(Bs, LDT, wn + 16, kk);
#pragma unroll
      for (int mi = 0; mi < 4; ++mi) {
        v16h a = lds_frag_a(As, LDT, wm + mi * 16, kk);
        acc[mi][0] = wmma16(a, b0, acc[mi][0]);
        acc[mi][1] = wmma16(a, b1, acc[mi][1]);
      }
    }
    __syncthreads();
  }
  const int g = lane >> 4, cl = lane & 15;
#pragma unroll
  for (int mi = 0; mi < 4; ++mi)
#pragma unroll
    for (int ni = 0; ni < 2; ++ni) {
      int gcol = bn + wn + ni * 16 + cl;
      float bv = bias[gcol];
#pragma unroll
      for (int v = 0; v < 8; ++v) {
        int grow = bm + wm + mi * 16 + g * 8 + v;
        Out[(size_t)grow * E_ + gcol] = acc[mi][ni][v] + bv;
      }
    }
}

// ---------------------------------------------------------------------------
extern "C" void kernel_launch(void* const* d_in, const int* in_sizes, int n_in,
                              void* d_out, int out_size, void* d_ws,
                              size_t ws_size, hipStream_t stream) {
  const float* Q        = (const float*)d_in[0];
  const float* K        = (const float*)d_in[1];
  const float* V        = (const float*)d_in[2];
  const float* q_w      = (const float*)d_in[3];
  const float* k_w      = (const float*)d_in[4];
  const float* v_w      = (const float*)d_in[5];
  const float* out_w    = (const float*)d_in[6];
  const float* in_bias  = (const float*)d_in[7];
  const float* out_bias = (const float*)d_in[8];
  const float* attn_W   = (const float*)d_in[9];
  float* out = (float*)d_out;
  char* ws = (char*)d_ws;

  const size_t HEADBUF = (size_t)H_ * NB_ * L_ * D_ * sizeof(_Float16);  // 8 MB
  _Float16* Qh  = (_Float16*)(ws);
  _Float16* Kh  = (_Float16*)(ws + 1 * HEADBUF);
  _Float16* Vt  = (_Float16*)(ws + 2 * HEADBUF);
  _Float16* Qt  = (_Float16*)(ws + 3 * HEADBUF);
  _Float16* O16 = (_Float16*)(ws + 4 * HEADBUF);
  float* Mrow   = (float*)(ws + 5 * HEADBUF);
  float* Zrow   = (float*)(ws + 5 * HEADBUF + ((size_t)1 << 20));

  dim3 blk(256);
  k_inproj<<<dim3(MR_ / 128, E_ / 128), blk, 0, stream>>>(Q, q_w, in_bias, Qh, L_, 0);
  k_inproj<<<dim3(MR_ / 128, E_ / 128), blk, 0, stream>>>(K, k_w, in_bias + E_, Kh, S_, 0);
  k_inproj<<<dim3(MR_ / 128, E_ / 128), blk, 0, stream>>>(V, v_w, in_bias + 2 * E_, Vt, S_, 1);
  k_qt<<<dim3(MR_ / 128, H_), blk, 0, stream>>>(Qh, attn_W, Qt);
  k_stats<<<dim3(L_ / 64, NB_, H_), blk, 0, stream>>>(Qt, Kh, Mrow, Zrow);
  k_pv<<<dim3(L_ / 64, NB_, H_), blk, 0, stream>>>(Qt, Kh, Vt, Mrow, Zrow, O16);
  k_wmap<<<dim3(S_ / 128, L_ / 64, NB_), blk, 0, stream>>>(
      Qt, Kh, Mrow, Zrow, out + (size_t)L_ * NB_ * E_);
  k_outproj<<<dim3(MR_ / 128, E_ / 128), blk, 0, stream>>>(O16, out_w, out_bias, out);
}
